// ViewEncoder_36764920054543
// MI455X (gfx1250) — compile-verified, matrix operand order checked by reading, not run on hardware
//
#include <hip/hip_runtime.h>

// ---------------------------------------------------------------------------
// 2-layer GCN forward for MI455X (gfx1250, wave32).
//   agg1 = SpMM(A, x); h = relu(agg1@W1+b1); agg2 = SpMM(A, h); z = agg2@W2+b2
// SpMM : wave-per-edge, float4 per lane, hardware global_atomic_add_f32
//        (accumulator fits in 192MB L2 -> atomics stay on-chip).
// GEMM : bf16 WMMA 16x16x32. W is repacked once into fragment-native bf16
//        (1 x 32B global load per K-step); the block's A tile is staged in
//        LDS pre-converted to bf16 in fragment layout (1 x 32B ds load per
//        K-step), shared by all 8 waves. Fused bias (+relu) on store.
// ---------------------------------------------------------------------------

#define N_NODES 100000
#define N_EDGES 1600000
#define DIM     128   // IN_DIM == HID_DIM == 128

typedef __bf16 v16bf __attribute__((ext_vector_type(16)));
typedef __bf16 v8bf  __attribute__((ext_vector_type(8)));
typedef float  v8f   __attribute__((ext_vector_type(8)));

__device__ __forceinline__ __bf16 f2bf(float f) { return (__bf16)f; }

// ---------------------------------------------------------------------------
// SpMM: one wave (32 lanes) per edge; lane l handles features [4l, 4l+4).
//   agg[dst][*] += val * x[src][*]
// ---------------------------------------------------------------------------
__global__ void spmm_kernel(const float* __restrict__ x,
                            const int*   __restrict__ esrc,
                            const int*   __restrict__ edst,
                            const float* __restrict__ evals,
                            float*       __restrict__ agg)
{
    const int lane = threadIdx.x & 31;
    const int e    = blockIdx.x * (blockDim.x >> 5) + (threadIdx.x >> 5);
    if (e >= N_EDGES) return;

    const int   s = esrc[e];
    const int   d = edst[e];
    const float v = evals[e];

    const float4 xv = *(const float4*)(&x[(size_t)s * DIM + lane * 4]);
    float* dp = &agg[(size_t)d * DIM + lane * 4];
    unsafeAtomicAdd(dp + 0, v * xv.x);
    unsafeAtomicAdd(dp + 1, v * xv.y);
    unsafeAtomicAdd(dp + 2, v * xv.z);
    unsafeAtomicAdd(dp + 3, v * xv.w);
}

// ---------------------------------------------------------------------------
// Fragment k-mapping (CDNA5 ISA 7.12.2, 16-bit A/B, wave32):
//   packed reg v (0..7) holds K pair { (v&4)<<2 | hi<<3 | (v&3)<<1 , +1 }.
//   Equivalently, fragment element j = half*8 + i  (half = j>=8)
//   maps to k = kTile*32 + half*16 + hi*8 + i  -> contiguous runs of 8.
// ---------------------------------------------------------------------------

// Repack W[K=128][N] (row-major fp32) -> Wp[kTile][nTile][lane][16] bf16.
// One thread per (kTile, nTile, lane, half) unit = 8 bf16 = 16 bytes.
template <int N>
__global__ void repack_w_kernel(const float* __restrict__ W,
                                __bf16*      __restrict__ Wp)
{
    const int TN    = N / 16;
    const int units = 4 * TN * 32 * 2;
    const int u = blockIdx.x * blockDim.x + threadIdx.x;
    if (u >= units) return;

    const int half  = u & 1;
    const int lane  = (u >> 1) & 31;
    const int nTile = (u >> 6) % TN;
    const int kTile = (u >> 6) / TN;
    const int hi    = lane >> 4;
    const int n     = nTile * 16 + (lane & 15);
    const int kb    = kTile * 32 + half * 16 + hi * 8;

    v8bf val;
#pragma unroll
    for (int i = 0; i < 8; ++i)
        val[i] = f2bf(W[(size_t)(kb + i) * N + n]);

    *(v8bf*)(&Wp[(((size_t)kTile * TN + nTile) * 32 + lane) * 16 + half * 8]) = val;
}

// ---------------------------------------------------------------------------
// GEMM: out[M,N] = act(A[M,128] @ W + bias). 256 threads = 8 waves.
// Wave (waveM, waveN) computes rows [mBase,mBase+16) x cols [nBase,nBase+16).
// ---------------------------------------------------------------------------
template <int N, int TILES_N, int ROW_TILES>
__global__ void gemm_kernel(const float*  __restrict__ A,
                            const __bf16* __restrict__ Wp,
                            const float*  __restrict__ bias,
                            float*        __restrict__ out,
                            int do_relu)
{
    __shared__ __bf16 ldsA[ROW_TILES][4][32][16];   // fragment-native A tiles

    const int lane   = threadIdx.x & 31;
    const int mBlock = blockIdx.x * (ROW_TILES * 16);

    // ---- cooperative stage: A tile(s) -> bf16 fragment layout in LDS ----
    for (int u = threadIdx.x; u < ROW_TILES * 256; u += 256) {
        const int rt    = u >> 8;
        const int t     = u & 255;
        const int kTile = t >> 6;          // 0..3
        const int s     = t & 63;
        const int sl    = s >> 1;          // staging lane 0..31
        const int half  = s & 1;
        const int shi   = sl >> 4;
        const int row   = mBlock + rt * 16 + (sl & 15);
        const int kb    = kTile * 32 + half * 16 + shi * 8;

        const float4 f0 = *(const float4*)(&A[(size_t)row * DIM + kb]);
        const float4 f1 = *(const float4*)(&A[(size_t)row * DIM + kb + 4]);
        v8bf val;
        val[0] = f2bf(f0.x); val[1] = f2bf(f0.y);
        val[2] = f2bf(f0.z); val[3] = f2bf(f0.w);
        val[4] = f2bf(f1.x); val[5] = f2bf(f1.y);
        val[6] = f2bf(f1.z); val[7] = f2bf(f1.w);
        *(v8bf*)(&ldsA[rt][kTile][sl][half * 8]) = val;
    }
    __syncthreads();

    // ---- WMMA main loop: 1 ds 32B + 1 global 32B + 1 wmma per K-step ----
    const int wave   = threadIdx.x >> 5;
    const int waveN  = wave % TILES_N;
    const int waveM  = wave / TILES_N;
    const int laneLo = lane & 15;
    const int hi     = lane >> 4;
    const int col    = waveN * 16 + laneLo;

    v8f c = {};
#pragma unroll
    for (int kTile = 0; kTile < 4; ++kTile) {
        const v16bf a = *(const v16bf*)(&ldsA[waveM][kTile][lane][0]);
        const v16bf b = *(const v16bf*)(
            &Wp[(((size_t)kTile * TILES_N + waveN) * 32 + lane) * 16]);
        c = __builtin_amdgcn_wmma_f32_16x16x32_bf16(
                /*neg_a=*/false, a, /*neg_b=*/false, b,
                /*c_mod=*/(short)0, c, /*reuse_a=*/false, /*reuse_b=*/false);
    }

    // ---- epilogue: bias (+relu), store fp32 ----
    const float bv    = bias[col];
    const int   mBase = mBlock + waveM * 16;
#pragma unroll
    for (int v = 0; v < 8; ++v) {
        float val = c[v] + bv;
        if (do_relu) val = fmaxf(val, 0.0f);
        out[(size_t)(mBase + (hi << 3) + v) * N + col] = val;
    }
}

// ---------------------------------------------------------------------------
extern "C" void kernel_launch(void* const* d_in, const int* in_sizes, int n_in,
                              void* d_out, int out_size, void* d_ws, size_t ws_size,
                              hipStream_t stream)
{
    const float* x     = (const float*)d_in[0];
    const int*   esrc  = (const int*)  d_in[1];
    const int*   edst  = (const int*)  d_in[2];
    const float* evals = (const float*)d_in[3];
    const float* W1    = (const float*)d_in[4];
    const float* b1    = (const float*)d_in[5];
    const float* W2    = (const float*)d_in[6];
    const float* b2    = (const float*)d_in[7];
    float*       out   = (float*)d_out;

    const size_t featElems = (size_t)N_NODES * DIM;       // 12.8M floats
    float*  agg = (float*)d_ws;                           // SpMM accumulator (reused)
    float*  h   = agg + featElems;                        // hidden activations
    __bf16* wp1 = (__bf16*)(h + featElems);               // 4*8*32*16 = 16384 bf16
    __bf16* wp2 = wp1 + 4 * 8 * 32 * 16;                  // 4*4*32*16 =  8192 bf16

    const int spmmBlocks = N_EDGES / 8;                   // 8 edges per 256-thr block

    // One-shot weight repacks (fragment-native bf16).
    repack_w_kernel<128><<<8, 256, 0, stream>>>(W1, wp1); // 2048 units
    repack_w_kernel<64><<<4, 256, 0, stream>>>(W2, wp2);  // 1024 units

    // Layer 1: agg = SpMM(A, x); h = relu(agg @ W1 + b1)
    hipMemsetAsync(agg, 0, featElems * sizeof(float), stream);
    spmm_kernel<<<spmmBlocks, 256, 0, stream>>>(x, esrc, edst, evals, agg);
    gemm_kernel<128, 8, 1><<<N_NODES / 16, 256, 0, stream>>>(agg, wp1, b1, h, 1);

    // Layer 2: agg = SpMM(A, h); out = agg @ W2 + b2
    hipMemsetAsync(agg, 0, featElems * sizeof(float), stream);
    spmm_kernel<<<spmmBlocks, 256, 0, stream>>>(h, esrc, edst, evals, agg);
    gemm_kernel<64, 4, 2><<<N_NODES / 32, 256, 0, stream>>>(agg, wp2, b2, out, 0);
}
